// _NeuronFusedSecondary_25409026523688
// MI455X (gfx1250) — compile-verified, matrix operand order checked by reading, not run on hardware
//
#include <hip/hip_runtime.h>
#include <hip/hip_bf16.h>

// ---------------------------------------------------------------------------
// Model constants (from reference)
// ---------------------------------------------------------------------------
#define BB   2
#define SS   512
#define LL   12
#define DIMC 1024
#define HH   16
#define HDC  64
#define DFFC 4096
#define VOC  2048
#define CHD  7
#define MM   (BB*SS)          // 1024 rows of activations

// ---------------------------------------------------------------------------
// Types
// ---------------------------------------------------------------------------
typedef __attribute__((ext_vector_type(16))) __bf16         v16bf;
typedef __attribute__((ext_vector_type(8)))  float          v8f;
typedef __attribute__((ext_vector_type(8)))  unsigned short us8;
typedef __attribute__((ext_vector_type(4)))  float          f4;
typedef __attribute__((ext_vector_type(2)))  unsigned int   u2;

// round-to-nearest-even f32 -> bf16, two at a time packed into one dword
__device__ __forceinline__ unsigned pack_bf16(float a, float b) {
  union { float f; unsigned u; } x, y; x.f = a; y.f = b;
  unsigned ra = x.u + 0x7FFFu + ((x.u >> 16) & 1u);
  unsigned rb = y.u + 0x7FFFu + ((y.u >> 16) & 1u);
  return (ra >> 16) | (rb & 0xFFFF0000u);
}

// 16x32 bf16 fragment (A layout; B identical on the transposed LDS tile).
// Lanes 0-15: K = [0..7]+[16..23]; lanes 16-31: K = [8..15]+[24..31]
//   -> kb = (lane>>4)*8, two contiguous 16B LDS reads (ds_load_b128).
__device__ __forceinline__ v16bf load_frag(const unsigned short* rowbase, int kb) {
  union { us8 h[2]; v16bf v; } u;
  u.h[0] = *(const us8*)(rowbase + kb);
  u.h[1] = *(const us8*)(rowbase + kb + 16);
  return u.v;
}

// ---------------------------------------------------------------------------
// Tiled bf16 WMMA GEMM: C[M,N] = A[M,K] @ W[K,N]   (f32 in memory)
// Block tile 128x128, K-step 32, 256 threads = 8 wave32.
// Wave grid 4x2; each wave owns a 32x64 slab -> acc[2][4], 8 WMMAs / K-stage.
// mode 0: C = A@W     mode 1: C = resid + A@W
// mode 2: per-head scatter to (B, CHD, S, N), head = blockIdx.z
// ---------------------------------------------------------------------------
#define BM 128
#define BN 128
#define BK 32
#define LDA 40   // padded LDS row stride (halves): 80B keeps 16B alignment and
                 // spreads the 16 fragment rows over 16 distinct bank groups

__global__ __launch_bounds__(256)
void gemm_bf16_wmma(const float* __restrict__ A, const float* __restrict__ Wt,
                    float* C, const float* resid,
                    int M, int N, int K, int mode)
{
  __shared__ __align__(16) unsigned short As[BM * LDA];  // As[m][k]
  __shared__ __align__(16) unsigned short Bs[BN * LDA];  // Bs[n][k] (W transposed)

  const int tid  = threadIdx.x;
  const int lane = tid & 31;
  const int wv   = tid >> 5;
  const int wr   = wv >> 1;           // 4 wave-rows of 32
  const int wc   = wv & 1;            // 2 wave-cols of 64
  const int bm   = blockIdx.y * BM;
  const int bn   = blockIdx.x * BN;

  const float* Wh = (mode == 2) ? (Wt + (size_t)blockIdx.z * K * N) : Wt;

  v8f acc[2][4];
#pragma unroll
  for (int mi = 0; mi < 2; ++mi)
#pragma unroll
    for (int ni = 0; ni < 4; ++ni) acc[mi][ni] = (v8f){};

  for (int k0 = 0; k0 < K; k0 += BK) {
    // ---- stage A tile (128x32): float4 load -> 2 packed dwords -> b64 store
#pragma unroll
    for (int i = 0; i < 4; ++i) {
      int slot = tid + i * 256;               // 1024 float4 slots
      int r = slot >> 3, c4 = (slot & 7) * 4;
      f4 v = *(const f4*)(A + (size_t)(bm + r) * K + k0 + c4);
      u2 p; p.x = pack_bf16(v.x, v.y); p.y = pack_bf16(v.z, v.w);
      *(u2*)(&As[r * LDA + c4]) = p;
    }
    // ---- stage W tile (32x128) transposed: 2x4 f32 patch -> 4 packed dwords
#pragma unroll
    for (int i = 0; i < 2; ++i) {
      int patch = tid + i * 256;              // 512 patches (2k x 4n)
      int k = (patch >> 5) * 2;
      int n = (patch & 31) * 4;
      const float* w0 = Wh + (size_t)(k0 + k) * N + bn + n;
      f4 a0 = *(const f4*)w0;
      f4 a1 = *(const f4*)(w0 + N);
#pragma unroll
      for (int j = 0; j < 4; ++j)
        *(unsigned*)(&Bs[(n + j) * LDA + k]) = pack_bf16(a0[j], a1[j]);
    }
    // hint next K-tile into cache while we compute (global_prefetch_b8)
    if (k0 + BK < K) {
      __builtin_prefetch(&A[(size_t)(bm + (tid >> 1)) * K + k0 + BK], 0, 0);
      __builtin_prefetch(&Wh[(size_t)(k0 + BK + (tid >> 3)) * N + bn], 0, 0);
    }
    __syncthreads();

    const int kb = (lane >> 4) * 8;
    const int rl = lane & 15;
    v16bf af[2], bfr[4];
#pragma unroll
    for (int mi = 0; mi < 2; ++mi)
      af[mi] = load_frag(&As[(wr * 32 + mi * 16 + rl) * LDA], kb);
#pragma unroll
    for (int ni = 0; ni < 4; ++ni)
      bfr[ni] = load_frag(&Bs[(wc * 64 + ni * 16 + rl) * LDA], kb);
#pragma unroll
    for (int mi = 0; mi < 2; ++mi)
#pragma unroll
      for (int ni = 0; ni < 4; ++ni)
        acc[mi][ni] = __builtin_amdgcn_wmma_f32_16x16x32_bf16(
            false, af[mi], false, bfr[ni], (short)0, acc[mi][ni], false, false);
    __syncthreads();
  }

  // ---- epilogue: C/D layout — VGPR r holds M = r (+8 for lanes 16-31) ----
  const int rl = lane & 15;
  const int mo = (lane >> 4) * 8;
#pragma unroll
  for (int mi = 0; mi < 2; ++mi) {
#pragma unroll
    for (int ni = 0; ni < 4; ++ni) {
#pragma unroll
      for (int r = 0; r < 8; ++r) {
        int m = bm + wr * 32 + mi * 16 + mo + r;
        int n = bn + wc * 64 + ni * 16 + rl;
        float v = acc[mi][ni][r];
        if (mode == 1) v += resid[(size_t)m * N + n];
        if (mode == 2) {
          int b = m / SS, s = m % SS;
          size_t ro = ((size_t)(b * CHD + blockIdx.z) * SS + s) * (size_t)N;
          C[ro + n] = v;
        } else {
          C[(size_t)m * N + n] = v;
        }
      }
    }
  }
}

// ---------------------------------------------------------------------------
// concat: out[m, 0:D] = a[m], out[m, D:2D] = b[m]
// ---------------------------------------------------------------------------
__global__ void concat2_kernel(const float* __restrict__ a, const float* __restrict__ b,
                               float* __restrict__ out, int total, int D)
{
  int i = blockIdx.x * blockDim.x + threadIdx.x;
  if (i >= total) return;
  int row = i / (2 * D), c = i % (2 * D);
  out[i] = (c < D) ? a[(size_t)row * D + c] : b[(size_t)row * D + (c - D)];
}

// ---------------------------------------------------------------------------
// RMSNorm: one block per row (256 threads, 8 wave32), wave-shuffle reduce
// ---------------------------------------------------------------------------
__global__ __launch_bounds__(256)
void rmsnorm_kernel(const float* __restrict__ h, const float* __restrict__ w,
                    float* __restrict__ out, int D)
{
  __shared__ float red[8];
  const int row = blockIdx.x;
  const float* hr = h + (size_t)row * D;
  float s = 0.f;
  for (int i = threadIdx.x; i < D; i += blockDim.x) { float v = hr[i]; s += v * v; }
  for (int off = 16; off; off >>= 1) s += __shfl_down(s, off, 32);
  int lane = threadIdx.x & 31, wv = threadIdx.x >> 5;
  if (lane == 0) red[wv] = s;
  __syncthreads();
  if (wv == 0) {
    s = (lane < 8) ? red[lane] : 0.f;
    for (int off = 4; off; off >>= 1) s += __shfl_down(s, off, 32);
    if (lane == 0) red[0] = s;
  }
  __syncthreads();
  float inv = rsqrtf(red[0] / (float)D + 1e-5f);
  for (int i = threadIdx.x; i < D; i += blockDim.x)
    out[(size_t)row * D + i] = hr[i] * inv * w[i];
}

// ---------------------------------------------------------------------------
// RoPE (position = s); each thread rotates one (d, d+32) pair — no hazards.
// ---------------------------------------------------------------------------
__global__ void rope_kernel(float* __restrict__ x, int npairs)
{
  int i = blockIdx.x * blockDim.x + threadIdx.x;
  if (i >= npairs) return;
  int row = i / (DIMC / 2);          // b*S + s
  int p   = i % (DIMC / 2);
  int hh  = p >> 5;                  // head
  int fr  = p & 31;                  // frequency index
  int s   = row % SS;
  float ang = (float)s * __expf(-0.28782313662f * (float)fr); // 10000^(-fr/32)
  float c = __cosf(ang), sn = __sinf(ang);
  size_t i0 = (size_t)row * DIMC + hh * HDC + fr;
  size_t i1 = i0 + 32;
  float x0 = x[i0], x1 = x[i1];
  x[i0] = x0 * c - x1 * sn;
  x[i1] = x1 * c + x0 * sn;
}

// ---------------------------------------------------------------------------
// Causal attention, one block (128 threads) per (b, h, query-row).
// ---------------------------------------------------------------------------
__global__ __launch_bounds__(128)
void attention_kernel(const float* __restrict__ q, const float* __restrict__ k,
                      const float* __restrict__ v, float* __restrict__ o)
{
  const int idx = blockIdx.x;
  const int sq  = idx % SS;
  const int h   = (idx / SS) % HH;
  const int b   = idx / (SS * HH);
  const int tid = threadIdx.x;
  const int lane = tid & 31, wv = tid >> 5;

  __shared__ __align__(16) float qs[HDC];
  __shared__ float sc[SS];
  __shared__ float red[4];

  const size_t qoff = ((size_t)(b * SS + sq)) * DIMC + h * HDC;
  if (tid < HDC) qs[tid] = q[qoff + tid];
  __syncthreads();

  const int T = sq + 1;
  const f4* q4 = (const f4*)qs;
  for (int t = tid; t < T; t += 128) {
    const f4* k4 = (const f4*)(k + ((size_t)(b * SS + t)) * DIMC + h * HDC);
    float d = 0.f;
#pragma unroll
    for (int e = 0; e < HDC / 4; ++e) {
      f4 a = q4[e], bb = k4[e];
      d += a.x * bb.x + a.y * bb.y + a.z * bb.z + a.w * bb.w;
    }
    sc[t] = d * 0.125f;
  }
  __syncthreads();

  // --- block max ---
  float mx = -1e30f;
  for (int t = tid; t < T; t += 128) mx = fmaxf(mx, sc[t]);
  for (int off = 16; off; off >>= 1) mx = fmaxf(mx, __shfl_down(mx, off, 32));
  if (lane == 0) red[wv] = mx;
  __syncthreads();
  mx = fmaxf(fmaxf(red[0], red[1]), fmaxf(red[2], red[3]));
  __syncthreads();

  // --- exp + block sum ---
  float sum = 0.f;
  for (int t = tid; t < T; t += 128) { float e = __expf(sc[t] - mx); sc[t] = e; sum += e; }
  for (int off = 16; off; off >>= 1) sum += __shfl_down(sum, off, 32);
  if (lane == 0) red[wv] = sum;
  __syncthreads();
  sum = red[0] + red[1] + red[2] + red[3];
  const float inv = 1.f / sum;
  __syncthreads();

  // --- weighted sum of V (threads over head-dim) ---
  for (int d = tid; d < HDC; d += 128) {
    float accv = 0.f;
    const float* vb = v + ((size_t)(b * SS)) * DIMC + h * HDC + d;
    for (int t = 0; t < T; ++t) accv += sc[t] * vb[(size_t)t * DIMC];
    o[qoff + d] = accv * inv;
  }
}

// ---------------------------------------------------------------------------
// SiLU(gate) * up  (in-place into gate buffer)
// ---------------------------------------------------------------------------
__global__ void silu_mul_kernel(float* __restrict__ g, const float* __restrict__ u, int n)
{
  int i = blockIdx.x * blockDim.x + threadIdx.x;
  if (i >= n) return;
  float x = g[i];
  g[i] = (x / (1.f + __expf(-x))) * u[i];
}

// ---------------------------------------------------------------------------
// Host orchestration
// ---------------------------------------------------------------------------
static inline void launch_gemm(const float* A, const float* W, float* C,
                               const float* resid, int M, int N, int K,
                               int mode, int zdim, hipStream_t s)
{
  dim3 grid(N / BN, M / BM, zdim);
  gemm_bf16_wmma<<<grid, 256, 0, s>>>(A, W, C, resid, M, N, K, mode);
}

extern "C" void kernel_launch(void* const* d_in, const int* in_sizes, int n_in,
                              void* d_out, int out_size, void* d_ws, size_t ws_size,
                              hipStream_t stream)
{
  (void)in_sizes; (void)n_in; (void)out_size; (void)ws_size;

  const float* fused    = (const float*)d_in[0];
  const float* primary  = (const float*)d_in[1];
  const float* w_bridge = (const float*)d_in[3];
  const float* ln1_w    = (const float*)d_in[4];
  const float* wq       = (const float*)d_in[5];
  const float* wk       = (const float*)d_in[6];
  const float* wvv      = (const float*)d_in[7];
  const float* wo       = (const float*)d_in[8];
  const float* ln2_w    = (const float*)d_in[9];
  const float* w_gate   = (const float*)d_in[10];
  const float* w_up     = (const float*)d_in[11];
  const float* w_down   = (const float*)d_in[12];
  const float* norm_w   = (const float*)d_in[13];
  const float* w_out    = (const float*)d_in[14];
  float* out = (float*)d_out;

  // workspace layout (f32 elements) — 16M floats = 64 MB total
  float* ws  = (float*)d_ws;
  float* cat = ws;                          // M * 2D
  float* hb  = cat + (size_t)MM * 2 * DIMC;
  float* xb  = hb  + (size_t)MM * DIMC;
  float* qb  = xb  + (size_t)MM * DIMC;
  float* kb  = qb  + (size_t)MM * DIMC;
  float* vb  = kb  + (size_t)MM * DIMC;
  float* ob  = vb  + (size_t)MM * DIMC;
  float* gb  = ob  + (size_t)MM * DIMC;     // M * DFF
  float* ub  = gb  + (size_t)MM * DFFC;     // M * DFF

  const int catN = MM * 2 * DIMC;
  concat2_kernel<<<(catN + 255) / 256, 256, 0, stream>>>(fused, primary, cat, catN, DIMC);

  // h = cat @ w_bridge   [1024 x 2048] @ [2048 x 1024]
  launch_gemm(cat, w_bridge, hb, nullptr, MM, DIMC, 2 * DIMC, 0, 1, stream);

  const int npairs = MM * DIMC / 2;
  const int ffN    = MM * DFFC;

  for (int i = 0; i < LL; ++i) {
    const size_t wOff = (size_t)i * DIMC * DIMC;
    const size_t fOff = (size_t)i * DIMC * DFFC;

    rmsnorm_kernel<<<MM, 256, 0, stream>>>(hb, ln1_w + (size_t)i * DIMC, xb, DIMC);

    launch_gemm(xb, wq  + wOff, qb, nullptr, MM, DIMC, DIMC, 0, 1, stream);
    launch_gemm(xb, wk  + wOff, kb, nullptr, MM, DIMC, DIMC, 0, 1, stream);
    launch_gemm(xb, wvv + wOff, vb, nullptr, MM, DIMC, DIMC, 0, 1, stream);

    rope_kernel<<<(npairs + 255) / 256, 256, 0, stream>>>(qb, npairs);
    rope_kernel<<<(npairs + 255) / 256, 256, 0, stream>>>(kb, npairs);

    attention_kernel<<<BB * HH * SS, 128, 0, stream>>>(qb, kb, vb, ob);

    // h = h + o @ wo
    launch_gemm(ob, wo + wOff, hb, hb, MM, DIMC, DIMC, 1, 1, stream);

    rmsnorm_kernel<<<MM, 256, 0, stream>>>(hb, ln2_w + (size_t)i * DIMC, xb, DIMC);

    launch_gemm(xb, w_gate + fOff, gb, nullptr, MM, DFFC, DIMC, 0, 1, stream);
    launch_gemm(xb, w_up   + fOff, ub, nullptr, MM, DFFC, DIMC, 0, 1, stream);

    silu_mul_kernel<<<(ffN + 255) / 256, 256, 0, stream>>>(gb, ub, ffN);

    // h = h + (silu(g)*u) @ w_down
    launch_gemm(gb, w_down + fOff, hb, hb, MM, DIMC, DFFC, 1, 1, stream);
  }

  rmsnorm_kernel<<<MM, 256, 0, stream>>>(hb, norm_w, xb, DIMC);

  // out[b,k,s,v] = xb @ w_out[k]  — head in blockIdx.z, scatter epilogue
  launch_gemm(xb, w_out, out, nullptr, MM, VOC, DIMC, 2, CHD, stream);
}
